// RelLearnableDecoderLayer_76965813944873
// MI455X (gfx1250) — matrix operand order repaired and checked by reading, not compiled, not run on hardware
//
#include <hip/hip_runtime.h>

typedef __attribute__((ext_vector_type(16))) __bf16 v16bf;
typedef __attribute__((ext_vector_type(8)))  float  v8f;
typedef __attribute__((ext_vector_type(4)))  unsigned int u32x4;
typedef __attribute__((ext_vector_type(4)))  float  f32x4;

union Frag  { v16bf f; u32x4 u[2]; };   // 16 bf16 = 32 B
union Pack8 { u32x4 u; __bf16 h[8]; };  // 8 bf16 = 16 B

#define QLEN 1024
#define BSZ 4
#define DMODEL 1024
#define NHEAD 16
#define DHEAD 64
#define DINNER 2048
#define NTOK (QLEN*BSZ)            // 4096 token rows (i*4+b)
#define ATTNSCALE 0.125f

// ---------------------------------------------------------------------------
// fp32 -> bf16 conversion, 8 elements per thread (b128 in/out)
// ---------------------------------------------------------------------------
__global__ void k_cvt_bf16x8(const float* __restrict__ x, __bf16* __restrict__ y, long long n8) {
  long long i = (long long)blockIdx.x * 256 + threadIdx.x;
  if (i >= n8) return;
  const f32x4* xv = (const f32x4*)x;
  f32x4 a = xv[2 * i], b = xv[2 * i + 1];
  Pack8 pk;
  #pragma unroll
  for (int j = 0; j < 4; ++j) { pk.h[j] = (__bf16)a[j]; pk.h[4 + j] = (__bf16)b[j]; }
  ((u32x4*)y)[i] = pk.u;
}

__global__ void k_zero_u32(unsigned* p, int n) {
  if (threadIdx.x < (unsigned)n) p[threadIdx.x] = 0u;
}

// ---------------------------------------------------------------------------
// Generic batched WMMA GEMM:  C[z] = A[z] (MxK row major) * W[z]^T (W is NxK)
// One 16x16 tile per wave, 8 waves/block = 128 N-columns, K unrolled by 64.
// z-dependent base offsets: off = (z/div)*sA + (z%div)*sB.
// mode 0: store fp32 C
// mode 2: store bf16 Cb + per-column bias  bias[(z/bdiv)*bsA + col*bcs]
// mode 3: score epilogue: (acc + B2[z][row, 1023+col-row])*scale, col>row -> -1e30
// ---------------------------------------------------------------------------
struct GemmP {
  const __bf16* A; int Adiv; long long AsA, AsB; int Arow;
  const __bf16* W; int Wdiv; long long WsA, WsB; int Wrow;
  float* C; __bf16* Cb; int Cdiv; long long CsA, CsB; int Crow;
  int M, N, K;
  int mode;
  const float* bias; int bdiv; long long bsA; int bcs;
  const __bf16* B2; long long B2batch; int B2row;
  float scale;
};

__global__ void k_gemm_wmma(GemmP p) {
  const int z = blockIdx.z;
  const long long zoA = (long long)(z / p.Adiv) * p.AsA + (long long)(z % p.Adiv) * p.AsB;
  const long long zoW = (long long)(z / p.Wdiv) * p.WsA + (long long)(z % p.Wdiv) * p.WsB;
  const long long zoC = (long long)(z / p.Cdiv) * p.CsA + (long long)(z % p.Cdiv) * p.CsB;
  const __bf16* A = p.A + zoA;
  const __bf16* W = p.W + zoW;

  const int wave = threadIdx.x >> 5;
  const int lane = threadIdx.x & 31;
  const int l15 = lane & 15, lhi = lane >> 4;
  const int row0 = blockIdx.y * 16;
  const int col0 = (blockIdx.x * 8 + wave) * 16;
  const bool active = (col0 < p.N);

  __shared__ u32x4 As4[128];          // 16 rows x 64 bf16 = 2 KB
  v8f acc = {};
  const int aU = l15 * 8 + lhi * 2;   // lane's A-frag base in u32x4 units (kc=0)

  for (int kk = 0; kk < p.K; kk += 64) {
    __syncthreads();
    if (threadIdx.x < 128) {          // one b128 per thread stages the A chunk
      int r = threadIdx.x >> 3, c8 = threadIdx.x & 7;
      As4[threadIdx.x] = *(const u32x4*)(A + (long long)(row0 + r) * p.Arow + kk + c8 * 8);
    }
    __syncthreads();
    if (active) {
      const __bf16* wrow = W + (long long)(col0 + l15) * p.Wrow + kk + lhi * 16;
      Frag af, bfr;
      // k sub-chunk 0
      af.u[0] = As4[aU];     af.u[1] = As4[aU + 1];
      bfr.u[0] = ((const u32x4*)wrow)[0];
      bfr.u[1] = ((const u32x4*)wrow)[1];
      acc = __builtin_amdgcn_wmma_f32_16x16x32_bf16(false, af.f, false, bfr.f,
                                                    (short)0, acc, false, false);
      // k sub-chunk 1 (+32)
      af.u[0] = As4[aU + 4]; af.u[1] = As4[aU + 5];
      bfr.u[0] = ((const u32x4*)(wrow + 32))[0];
      bfr.u[1] = ((const u32x4*)(wrow + 32))[1];
      acc = __builtin_amdgcn_wmma_f32_16x16x32_bf16(false, af.f, false, bfr.f,
                                                    (short)0, acc, false, false);
      if (kk + 64 < p.K) __builtin_prefetch(wrow + 64, 0, 1);
    }
  }
  if (!active) return;

  const int ccol = col0 + l15;
  if (p.mode == 0) {
    float* C = p.C + zoC;
    #pragma unroll
    for (int v = 0; v < 8; ++v) {
      int rv = row0 + v + lhi * 8;
      C[(long long)rv * p.Crow + ccol] = acc[v];
    }
  } else if (p.mode == 2) {
    __bf16* Cb = p.Cb + zoC;
    const float* bias = p.bias + (long long)(z / p.bdiv) * p.bsA;
    float bv = bias[(long long)ccol * p.bcs];
    #pragma unroll
    for (int v = 0; v < 8; ++v) {
      int rv = row0 + v + lhi * 8;
      Cb[(long long)rv * p.Crow + ccol] = (__bf16)(acc[v] + bv);
    }
  } else { // mode 3: rel-shift gather + scale + causal mask
    __bf16* Cb = p.Cb + zoC;
    const __bf16* B2 = p.B2 + (long long)z * p.B2batch;
    #pragma unroll
    for (int v = 0; v < 8; ++v) {
      int i = row0 + v + lhi * 8;
      float out;
      if (ccol > i) {
        out = -1e30f;
      } else {
        float bd = (float)B2[(long long)i * p.B2row + (QLEN - 1 + ccol - i)];
        out = (acc[v] + bd) * p.scale;
      }
      Cb[(long long)i * p.Crow + ccol] = (__bf16)out;
    }
  }
}

// ---------------------------------------------------------------------------
// Split heads fp32 [4096,3072] -> per-(n,b) bf16: hq, hq+r_w_bias, hk, hv^T
// z = n*4 + b ; hq/hqr/hk: [z][1024][64], hvT: [z][64][1024]
// ---------------------------------------------------------------------------
__global__ void k_split_heads(const float* __restrict__ heads, const float* __restrict__ rwb,
                              __bf16* hq, __bf16* hqr, __bf16* hk, __bf16* hvT) {
  long long idx = (long long)blockIdx.x * 256 + threadIdx.x;  // over 4096*1024
  if (idx >= (long long)NTOK * DMODEL) return;
  int d = idx & 63;
  int n = (idx >> 6) & 15;
  int t = (int)(idx >> 10);
  int b = t & 3, i = t >> 2;
  int z = n * 4 + b;
  const float* hr = heads + (long long)t * (3 * DMODEL);
  float q = hr[n * 64 + d];
  float k = hr[DMODEL + n * 64 + d];
  float v = hr[2 * DMODEL + n * 64 + d];
  long long zo = (long long)z * (QLEN * DHEAD);
  hq [zo + i * 64 + d] = (__bf16)q;
  hqr[zo + i * 64 + d] = (__bf16)(q + rwb[n * 64 + d]);
  hk [zo + i * 64 + d] = (__bf16)k;
  hvT[zo + (long long)d * QLEN + i] = (__bf16)v;
}

// ---------------------------------------------------------------------------
// Row softmax over 1024 bf16, in place. One wave per row; each lane owns a
// contiguous 64 B span (4x b128 load/store).
// ---------------------------------------------------------------------------
__global__ void k_softmax_rows(__bf16* S) {
  long long row = (long long)blockIdx.x * 8 + (threadIdx.x >> 5);
  int lane = threadIdx.x & 31;
  u32x4* rp = (u32x4*)(S + row * QLEN) + lane * 4;
  union { u32x4 v[4]; __bf16 h[32]; } u;
  #pragma unroll
  for (int c = 0; c < 4; ++c) u.v[c] = rp[c];
  float vals[32];
  float m = -3.0e38f;
  #pragma unroll
  for (int c = 0; c < 32; ++c) { vals[c] = (float)u.h[c]; m = fmaxf(m, vals[c]); }
  #pragma unroll
  for (int o = 16; o; o >>= 1) m = fmaxf(m, __shfl_xor(m, o, 32));
  float s = 0.f;
  #pragma unroll
  for (int c = 0; c < 32; ++c) { vals[c] = __expf(vals[c] - m); s += vals[c]; }
  #pragma unroll
  for (int o = 16; o; o >>= 1) s += __shfl_xor(s, o, 32);
  float inv = 1.0f / s;
  #pragma unroll
  for (int c = 0; c < 32; ++c) u.h[c] = (__bf16)(vals[c] * inv);
  #pragma unroll
  for (int c = 0; c < 4; ++c) rp[c] = u.v[c];
}

// ---------------------------------------------------------------------------
// Residual + LayerNorm over D=1024 per token; optional bf16 mirror output
// ---------------------------------------------------------------------------
__device__ __forceinline__ float blk_sum256(float v, volatile float* sm) {
  #pragma unroll
  for (int o = 16; o; o >>= 1) v += __shfl_xor(v, o, 32);
  __syncthreads();
  if ((threadIdx.x & 31) == 0) sm[threadIdx.x >> 5] = v;
  __syncthreads();
  if (threadIdx.x == 0) { float t = 0.f; for (int i = 0; i < 8; ++i) t += sm[i]; sm[8] = t; }
  __syncthreads();
  return sm[8];
}

__global__ void k_resid_ln(const float* __restrict__ resid, const float* __restrict__ x,
                           const float* __restrict__ bias, const float* __restrict__ g,
                           const float* __restrict__ b, float* __restrict__ out32,
                           __bf16* __restrict__ out16) {
  __shared__ float sm[9];
  long long base = (long long)blockIdx.x * DMODEL;
  float loc[4];
  float s = 0.f;
  #pragma unroll
  for (int e = 0; e < 4; ++e) {
    int j = threadIdx.x + e * 256;
    float v = resid[base + j] + x[base + j] + (bias ? bias[j] : 0.f);
    loc[e] = v; s += v;
  }
  float mu = blk_sum256(s, sm) * (1.0f / DMODEL);
  float s2 = 0.f;
  #pragma unroll
  for (int e = 0; e < 4; ++e) { float d = loc[e] - mu; s2 += d * d; }
  float var = blk_sum256(s2, sm) * (1.0f / DMODEL);
  float inv = rsqrtf(var + 1e-5f);
  #pragma unroll
  for (int e = 0; e < 4; ++e) {
    int j = threadIdx.x + e * 256;
    float y = (loc[e] - mu) * inv * g[j] + b[j];
    out32[base + j] = y;
    if (out16) out16[base + j] = (__bf16)y;
  }
}

// xp[t][c] += bih[c] + bhh[c]
__global__ void k_xp_bias(float* xp, const float* bih, const float* bhh) {
  long long i = (long long)blockIdx.x * 256 + threadIdx.x;  // over 4096*8192
  int c = (int)(i & (4 * DINNER - 1));
  xp[i] += bih[c] + bhh[c];
}

// ---------------------------------------------------------------------------
// Persistent LSTM recurrence. 64 blocks x 256 threads; block bk owns 32 hidden
// units (128 gate columns). h (4x2048, padded to M=16) @ whh^T via WMMA;
// whh bf16 (32 MB) stays L2-resident across all 1024 steps. Device-scope
// sense-reversing barrier between timesteps.
// ---------------------------------------------------------------------------
__device__ __forceinline__ float sigm(float x) { return 1.0f / (1.0f + __expf(-x)); }

__global__ void k_lstm(const __bf16* __restrict__ whh, const float* __restrict__ xp,
                       __bf16* __restrict__ hglob, __bf16* __restrict__ hs,
                       unsigned* barcnt, unsigned* bargen) {
  const int bk = blockIdx.x;
  const int tid = threadIdx.x;
  const int wave = tid >> 5, lane = tid & 31, l15 = lane & 15, lhi = lane >> 4;
  const int hu0 = bk * 32;
  const int g = wave >> 1, sub = wave & 1;
  const int colbase = g * DINNER + hu0 + sub * 16;

  __shared__ u32x4 hbuf4[1024];   // 4 x 2048 bf16 = 16 KB
  __shared__ float gbuf[4 * 4 * 32];
  __shared__ float cbuf[4 * 32];
  if (tid < 128) cbuf[tid] = 0.f;
  const __bf16* hbuf = (const __bf16*)hbuf4;
  const u32x4 zero4 = {0u, 0u, 0u, 0u};

  for (int t = 0; t < QLEN; ++t) {
    if (t == 0) {
      #pragma unroll
      for (int e = 0; e < 4; ++e) hbuf4[tid + e * 256] = zero4;
    } else {
      const u32x4* hg4 = (const u32x4*)hglob;
      #pragma unroll
      for (int e = 0; e < 4; ++e) hbuf4[tid + e * 256] = hg4[tid + e * 256];
    }
    __syncthreads();

    v8f acc = {};
    const int r = l15;
    for (int kk = 0; kk < DINNER; kk += 64) {
      const __bf16* wrow = whh + (long long)(colbase + l15) * DINNER + kk + lhi * 16;
      Frag af, bfr;
      #pragma unroll
      for (int kc = 0; kc < 2; ++kc) {
        if (r < 4) {
          const u32x4* hb = hbuf4 + r * 256 + ((kk + kc * 32) >> 3) + lhi * 2;
          af.u[0] = hb[0]; af.u[1] = hb[1];
        } else {
          af.u[0] = zero4; af.u[1] = zero4;
        }
        const u32x4* wp = (const u32x4*)(wrow + kc * 32);
        bfr.u[0] = wp[0]; bfr.u[1] = wp[1];
        acc = __builtin_amdgcn_wmma_f32_16x16x32_bf16(false, af.f, false, bfr.f,
                                                      (short)0, acc, false, false);
      }
      if (kk + 64 < DINNER) __builtin_prefetch(wrow + 64, 0, 1);
    }

    if (lhi == 0) {  // batch rows 0..3 live in elements v=0..3 of lanes 0..15
      #pragma unroll
      for (int v = 0; v < 4; ++v) {
        float pre = acc[v] + xp[(long long)(t * 4 + v) * (4 * DINNER) + colbase + l15];
        gbuf[(g * 4 + v) * 32 + sub * 16 + l15] = pre;
      }
    }
    __syncthreads();

    if (tid < 128) {
      int b = tid >> 5, j = tid & 31;
      float gi = gbuf[(0 * 4 + b) * 32 + j];
      float gf = gbuf[(1 * 4 + b) * 32 + j];
      float gg = gbuf[(2 * 4 + b) * 32 + j];
      float go = gbuf[(3 * 4 + b) * 32 + j];
      float c = sigm(gf) * cbuf[b * 32 + j] + sigm(gi) * tanhf(gg);
      cbuf[b * 32 + j] = c;
      float h = sigm(go) * tanhf(c);
      hglob[b * DINNER + hu0 + j] = (__bf16)h;
      hs[(long long)(t * 4 + b) * DINNER + hu0 + j] = (__bf16)h;
    }
    __threadfence();
    __syncthreads();
    if (tid == 0) {  // device-wide barrier (64 resident blocks)
      unsigned g0 = atomicAdd(bargen, 0u);
      unsigned a = atomicAdd(barcnt, 1u);
      if (a == 63u) { atomicExch(barcnt, 0u); atomicAdd(bargen, 1u); }
      else { while (atomicAdd(bargen, 0u) == g0) __builtin_amdgcn_s_sleep(8); }
    }
    __syncthreads();
    __threadfence();
  }
}

// ---------------------------------------------------------------------------
extern "C" void kernel_launch(void* const* d_in, const int* in_sizes, int n_in,
                              void* d_out, int out_size, void* d_ws, size_t ws_size,
                              hipStream_t stream) {
  const float* dec_inp  = (const float*)d_in[0];
  const float* r_emb    = (const float*)d_in[1];
  const float* r_w_bias = (const float*)d_in[2];
  const float* r_bias   = (const float*)d_in[3];
  const float* qkv_w2   = (const float*)d_in[9];
  const float* o_w2     = (const float*)d_in[10];
  const float* ln2_g    = (const float*)d_in[11];
  const float* ln2_b    = (const float*)d_in[12];
  const float* lstm_wih = (const float*)d_in[21];
  const float* lstm_whh = (const float*)d_in[22];
  const float* lstm_bih = (const float*)d_in[23];
  const float* lstm_bhh = (const float*)d_in[24];
  const float* lstm_pw  = (const float*)d_in[25];
  const float* lstm_pb  = (const float*)d_in[26];
  const float* lstm_lng = (const float*)d_in[27];
  const float* lstm_lnb = (const float*)d_in[28];
  float* out = (float*)d_out;

  // ---- workspace layout (deterministic bump allocator, with reuse) ----
  char* w = (char*)d_ws;
  auto alloc = [&](size_t bytes) { char* p = w; w += (bytes + 255) & ~(size_t)255; return p; };
  __bf16* dec_b16  = (__bf16*)alloc((size_t)NTOK * DMODEL * 2);
  __bf16* qkv2_b16 = (__bf16*)alloc((size_t)3 * DMODEL * DMODEL * 2);
  __bf16* ow2_b16  = (__bf16*)alloc((size_t)DMODEL * DMODEL * 2);
  __bf16* wih_b16  = (__bf16*)alloc((size_t)4 * DINNER * DMODEL * 2);
  __bf16* whh_b16  = (__bf16*)alloc((size_t)4 * DINNER * DINNER * 2);
  __bf16* pw_b16   = (__bf16*)alloc((size_t)DMODEL * DINNER * 2);
  __bf16* remb_b16 = (__bf16*)alloc((size_t)QLEN * NHEAD * DHEAD * 2);
  __bf16* hq  = (__bf16*)alloc((size_t)64 * QLEN * DHEAD * 2);
  __bf16* hqr = (__bf16*)alloc((size_t)64 * QLEN * DHEAD * 2);
  __bf16* hk  = (__bf16*)alloc((size_t)64 * QLEN * DHEAD * 2);
  __bf16* hvT = (__bf16*)alloc((size_t)64 * QLEN * DHEAD * 2);
  char*  regB2 = alloc((size_t)64 * QLEN * QLEN * 2);   // B2 bf16
  char*  regS  = alloc((size_t)64 * QLEN * QLEN * 2);   // S/P bf16; reused as proj f32
  float* xp    = (float*)alloc((size_t)NTOK * 4 * DINNER * 4);
  float* o2f   = (float*)alloc((size_t)NTOK * DMODEL * 4);
  __bf16* o2b  = (__bf16*)alloc((size_t)NTOK * DMODEL * 2);
  __bf16* hs   = (__bf16*)alloc((size_t)NTOK * DINNER * 2);
  __bf16* hgl  = (__bf16*)alloc((size_t)4 * DINNER * 2);
  unsigned* bar = (unsigned*)alloc(256);
  char* regHeads = alloc((size_t)NTOK * 3 * DMODEL * 4); // heads f32; reused for vec/oproj
  if ((size_t)(w - (char*)d_ws) > ws_size) return;       // workspace too small

  __bf16* B2    = (__bf16*)regB2;
  __bf16* S     = (__bf16*)regS;
  float*  proj  = (float*)regS;                   // reuse after P consumed
  float*  heads = (float*)regHeads;
  float*  vecf  = (float*)regHeads;               // reuse after split
  __bf16* vecb  = (__bf16*)(regHeads + (size_t)NTOK * DMODEL * 4);
  float*  oproj = (float*)(regHeads + (size_t)NTOK * DMODEL * 4 + (size_t)NTOK * DMODEL * 2);

  auto cvt = [&](const float* src, __bf16* dst, long long n) {
    long long n8 = n / 8;
    k_cvt_bf16x8<<<(unsigned)((n8 + 255) / 256), 256, 0, stream>>>(src, dst, n8);
  };
  cvt(dec_inp,  dec_b16,  (long long)NTOK * DMODEL);
  cvt(qkv_w2,   qkv2_b16, 3LL * DMODEL * DMODEL);
  cvt(o_w2,     ow2_b16,  (long long)DMODEL * DMODEL);
  cvt(lstm_wih, wih_b16,  4LL * DINNER * DMODEL);
  cvt(lstm_whh, whh_b16,  4LL * DINNER * DINNER);
  cvt(lstm_pw,  pw_b16,   (long long)DMODEL * DINNER);
  cvt(r_emb,    remb_b16, (long long)QLEN * NHEAD * DHEAD);
  k_zero_u32<<<1, 32, 0, stream>>>(bar, 2);

  auto gemm = [&](GemmP p, int gz) {
    dim3 grid((p.N + 127) / 128, p.M / 16, gz);
    k_gemm_wmma<<<grid, 256, 0, stream>>>(p);
  };
  GemmP p{};

  // 1) heads = dec_inp @ qkv_w2^T   [4096 x 3072]
  p = {}; p.A = dec_b16; p.Adiv = 1; p.Arow = DMODEL;
  p.W = qkv2_b16; p.Wdiv = 1; p.Wrow = DMODEL;
  p.C = heads; p.Cdiv = 1; p.Crow = 3 * DMODEL;
  p.M = NTOK; p.N = 3 * DMODEL; p.K = DMODEL; p.mode = 0;
  gemm(p, 1);

  // 2) split heads
  k_split_heads<<<(NTOK * DMODEL) / 256, 256, 0, stream>>>(heads, r_w_bias, hq, hqr, hk, hvT);

  // 3) B2[z] = hq[z] @ r_emb[n]^T + r_bias[:,n]     (z = n*4+b)
  p = {}; p.A = hq; p.Adiv = 1; p.AsA = QLEN * DHEAD; p.Arow = DHEAD;
  p.W = remb_b16; p.Wdiv = 4; p.WsA = DHEAD; p.Wrow = NHEAD * DHEAD;
  p.Cb = B2; p.Cdiv = 1; p.CsA = (long long)QLEN * QLEN; p.Crow = QLEN;
  p.M = QLEN; p.N = QLEN; p.K = DHEAD; p.mode = 2;
  p.bias = r_bias; p.bdiv = 4; p.bsA = 1; p.bcs = NHEAD;
  gemm(p, 64);

  // 4) S[z] = ((hq+rwb)[z] @ hk[z]^T + shift(B2)) * scale, causal mask
  p = {}; p.A = hqr; p.Adiv = 1; p.AsA = QLEN * DHEAD; p.Arow = DHEAD;
  p.W = hk; p.Wdiv = 1; p.WsA = QLEN * DHEAD; p.Wrow = DHEAD;
  p.Cb = S; p.Cdiv = 1; p.CsA = (long long)QLEN * QLEN; p.Crow = QLEN;
  p.M = QLEN; p.N = QLEN; p.K = DHEAD; p.mode = 3;
  p.B2 = B2; p.B2batch = (long long)QLEN * QLEN; p.B2row = QLEN; p.scale = ATTNSCALE;
  gemm(p, 64);

  // 5) softmax rows (in place, S -> P)
  k_softmax_rows<<<(64 * QLEN) / 8, 256, 0, stream>>>(S);

  // 6) vec = P @ hv :  C[i, b*1024 + n*64 + d]
  p = {}; p.A = S; p.Adiv = 1; p.AsA = (long long)QLEN * QLEN; p.Arow = QLEN;
  p.W = hvT; p.Wdiv = 1; p.WsA = QLEN * DHEAD; p.Wrow = QLEN;
  p.C = vecf; p.Cdiv = 4; p.CsA = DHEAD; p.CsB = DMODEL; p.Crow = BSZ * DMODEL;
  p.M = QLEN; p.N = DHEAD; p.K = QLEN; p.mode = 0;
  gemm(p, 64);

  cvt(vecf, vecb, (long long)NTOK * DMODEL);

  // 7) oproj = vec @ o_w2^T ; o2 = LN(dec_inp + oproj)
  p = {}; p.A = vecb; p.Adiv = 1; p.Arow = DMODEL;
  p.W = ow2_b16; p.Wdiv = 1; p.Wrow = DMODEL;
  p.C = oproj; p.Cdiv = 1; p.Crow = DMODEL;
  p.M = NTOK; p.N = DMODEL; p.K = DMODEL; p.mode = 0;
  gemm(p, 1);
  k_resid_ln<<<NTOK, 256, 0, stream>>>(dec_inp, oproj, nullptr, ln2_g, ln2_b, o2f, o2b);

  // 8) xp = o2 @ lstm_wih^T + (bih + bhh)
  p = {}; p.A = o2b; p.Adiv = 1; p.Arow = DMODEL;
  p.W = wih_b16; p.Wdiv = 1; p.Wrow = DMODEL;
  p.C = xp; p.Cdiv = 1; p.Crow = 4 * DINNER;
  p.M = NTOK; p.N = 4 * DINNER; p.K = DMODEL; p.mode = 0;
  gemm(p, 1);
  k_xp_bias<<<(unsigned)(((long long)NTOK * 4 * DINNER) / 256), 256, 0, stream>>>(xp, lstm_bih, lstm_bhh);

  // 9) sequential LSTM over 1024 steps (persistent kernel, device barrier)
  k_lstm<<<64, 256, 0, stream>>>(whh_b16, xp, hgl, hs, bar, bar + 1);

  // 10) proj = hs @ lstm_pw^T ; out = LN(o2 + proj + pb)
  p = {}; p.A = hs; p.Adiv = 1; p.Arow = DINNER;
  p.W = pw_b16; p.Wdiv = 1; p.Wrow = DINNER;
  p.C = proj; p.Cdiv = 1; p.Crow = DMODEL;
  p.M = NTOK; p.N = DMODEL; p.K = DINNER; p.mode = 0;
  gemm(p, 1);
  k_resid_ln<<<NTOK, 256, 0, stream>>>(o2f, proj, lstm_pb, lstm_lng, lstm_lnb, out, nullptr);

  (void)in_sizes; (void)n_in; (void)out_size;
}
